// MultiHeadAttention_62285615726624
// MI455X (gfx1250) — matrix operand order repaired and checked by reading, not compiled, not run on hardware
//
#include <hip/hip_runtime.h>
#include <hip/hip_bf16.h>

// ---------------------------------------------------------------------------
// MHA forward for MI455X (gfx1250, wave32, WMMA).
// [QKV proj bf16-WMMA GEMM, double-buffered] -> [flash attention, async K tile]
// -> [out proj GEMM]
// ---------------------------------------------------------------------------

typedef __bf16 v16bf __attribute__((ext_vector_type(16)));
typedef __bf16 v8bf  __attribute__((ext_vector_type(8)));
typedef __bf16 v2bf  __attribute__((ext_vector_type(2)));
typedef float  v8f   __attribute__((ext_vector_type(8)));
typedef int    v4i   __attribute__((ext_vector_type(4)));

#define DEVI __device__ __forceinline__

static constexpr int Bb = 4;
static constexpr int Ss = 2048;
static constexpr int Dd = 1024;
static constexpr int Hh = 16;
static constexpr int DH = 64;
static constexpr int Mm = Bb * Ss;   // 8192 rows

// hardware f32 -> bf16 (RTNE) convert
DEVI __bf16 f2bf(float f) { return (__bf16)f; }

DEVI v8f wmma_bf16(v16bf a, v16bf b, v8f c) {
  return __builtin_amdgcn_wmma_f32_16x16x32_bf16(false, a, false, b, (short)0, c,
                                                 false, false);
}

// ---- async global->LDS copy (ASYNCcnt path), guarded ----------------------
#if __has_builtin(__builtin_amdgcn_global_load_async_to_lds_b128) && \
    __has_builtin(__builtin_amdgcn_s_wait_asynccnt)
#define HAS_ASYNC_LDS 1
typedef __attribute__((address_space(1))) v4i gv4i;  // global (__device__) int4
typedef __attribute__((address_space(3))) v4i lv4i;  // LDS (__shared__) int4
DEVI void async_copy16(const void* g, void* l) {
  __builtin_amdgcn_global_load_async_to_lds_b128((gv4i*)g, (lv4i*)l, 0, 0);
}
DEVI void async_wait0() { __builtin_amdgcn_s_wait_asynccnt(0); }
#else
#define HAS_ASYNC_LDS 0
#endif

// ---------------------------------------------------------------------------
// GEMM: C[M,N] = A[M,K] * W[K,N] + bias
//   A_BF16  : 0 = A fp32 (x), 1 = A bf16 (attention output)
//   OUT_MODE: 0 = bf16 scattered to [B,H,S,64], 1 = fp32 row-major [M,N]
// 128x128x32 tiles, 256 threads (8 waves, 4Mx2N), wave = 32x64 via 2x4 WMMA.
// Double-buffered LDS, software-pipelined global loads, packed b32 transpose
// staging for the B tile.
// ---------------------------------------------------------------------------
template <int A_BF16, int OUT_MODE>
__global__ __launch_bounds__(256) void gemm128(
    const void* __restrict__ Ain, const float* __restrict__ W,
    const float* __restrict__ bias, void* __restrict__ Out,
    int M, int N, int K) {
  __shared__ alignas(16) __bf16 lA[2][128 * 32];  // [m][k]
  __shared__ alignas(16) __bf16 lB[2][128 * 32];  // [n][k]

  const int tid  = threadIdx.x;
  const int bm   = blockIdx.y, bn = blockIdx.x;
  const int wave = tid >> 5, lane = tid & 31;
  const int half = lane >> 4, l16 = lane & 15;
  const int wm   = wave >> 1, wn = wave & 1;

  v8f acc[2][4] = {};

  const int arow = tid >> 1, acol = (tid & 1) * 16;  // A stage: 16 elems
  const int bk   = (tid >> 4) * 2;                   // B stage: k pair 0..30
  const int bn8  = (tid & 15) * 8;                   // B stage: n base 0..120
  const int nk = K / 32;

  v8bf a_st0, a_st1;
  float fb0[8], fb1[8];

  auto loadA = [&](int k0) {
    if (A_BF16) {
      const __bf16* ga = (const __bf16*)Ain + (size_t)(bm * 128 + arow) * K + k0 + acol;
      a_st0 = *(const v8bf*)ga;
      a_st1 = *(const v8bf*)(ga + 8);
    } else {
      const float* ga = (const float*)Ain + (size_t)(bm * 128 + arow) * K + k0 + acol;
      float fa[16];
      *(float4*)&fa[0]  = *(const float4*)(ga);
      *(float4*)&fa[4]  = *(const float4*)(ga + 4);
      *(float4*)&fa[8]  = *(const float4*)(ga + 8);
      *(float4*)&fa[12] = *(const float4*)(ga + 12);
#pragma unroll
      for (int j = 0; j < 8; ++j) a_st0[j] = f2bf(fa[j]);
#pragma unroll
      for (int j = 0; j < 8; ++j) a_st1[j] = f2bf(fa[8 + j]);
    }
  };
  auto loadB = [&](int k0) {
    const float* gw = W + (size_t)(k0 + bk) * N + bn * 128 + bn8;
    *(float4*)&fb0[0] = *(const float4*)(gw);
    *(float4*)&fb0[4] = *(const float4*)(gw + 4);
    *(float4*)&fb1[0] = *(const float4*)(gw + N);
    *(float4*)&fb1[4] = *(const float4*)(gw + N + 4);
  };
  auto storeA = [&](int bsel) {
    *(v8bf*)&lA[bsel][arow * 32 + acol]     = a_st0;
    *(v8bf*)&lA[bsel][arow * 32 + acol + 8] = a_st1;
  };
  auto storeB = [&](int bsel) {
#pragma unroll
    for (int j = 0; j < 8; ++j) {
      v2bf p;
      p[0] = f2bf(fb0[j]);
      p[1] = f2bf(fb1[j]);
      *(v2bf*)&lB[bsel][(bn8 + j) * 32 + bk] = p;  // packed b32 store
    }
  };

  // ---- prologue: tile 0 ----
#if HAS_ASYNC_LDS
  if (A_BF16) {
    const __bf16* ga = (const __bf16*)Ain + (size_t)(bm * 128 + arow) * K + acol;
    async_copy16(ga,     &lA[0][arow * 32 + acol]);
    async_copy16(ga + 8, &lA[0][arow * 32 + acol + 8]);
  } else {
    loadA(0); storeA(0);
  }
#else
  loadA(0); storeA(0);
#endif
  loadB(0); storeB(0);
#if HAS_ASYNC_LDS
  if (A_BF16) async_wait0();
#endif
  __syncthreads();

  int buf = 0;
  for (int kt = 0; kt < nk; ++kt) {
    // ---- issue next tile's loads while computing this one ----
    if (kt + 1 < nk) {
      const int k0n = (kt + 1) * 32;
#if HAS_ASYNC_LDS
      if (A_BF16) {
        const __bf16* ga = (const __bf16*)Ain + (size_t)(bm * 128 + arow) * K + k0n + acol;
        async_copy16(ga,     &lA[buf ^ 1][arow * 32 + acol]);
        async_copy16(ga + 8, &lA[buf ^ 1][arow * 32 + acol + 8]);
      } else {
        loadA(k0n);
      }
#else
      loadA(k0n);
#endif
      loadB(k0n);
      if (kt + 2 < nk) {  // deep prefetch (global_prefetch_b8)
        __builtin_prefetch(W + (size_t)((kt + 2) * 32 + bk) * N + bn * 128 + bn8, 0, 0);
      }
    }

    // ---- fragments per ISA layouts ----
    v16bf afr[2];
#pragma unroll
    for (int i = 0; i < 2; ++i) {
      const __bf16* p = &lA[buf][(wm * 32 + i * 16 + l16) * 32 + half * 8];
      ((v8bf*)&afr[i])[0] = *(const v8bf*)p;
      ((v8bf*)&afr[i])[1] = *(const v8bf*)(p + 16);
    }
    v16bf bfr[4];
#pragma unroll
    for (int j = 0; j < 4; ++j) {
      const __bf16* p = &lB[buf][(wn * 64 + j * 16 + l16) * 32 + half * 8];
      ((v8bf*)&bfr[j])[0] = *(const v8bf*)p;
      ((v8bf*)&bfr[j])[1] = *(const v8bf*)(p + 16);
    }
#pragma unroll
    for (int i = 0; i < 2; ++i)
#pragma unroll
      for (int j = 0; j < 4; ++j)
        acc[i][j] = wmma_bf16(afr[i], bfr[j], acc[i][j]);

    // ---- commit next tile to the other buffer ----
    if (kt + 1 < nk) {
#if HAS_ASYNC_LDS
      if (!A_BF16) storeA(buf ^ 1);
#else
      storeA(buf ^ 1);
#endif
      storeB(buf ^ 1);
#if HAS_ASYNC_LDS
      if (A_BF16) async_wait0();
#endif
      __syncthreads();
      buf ^= 1;
    }
  }

  // ---- epilogue: C layout (col = l16, rows r + 8*half per VGPR) ----
#pragma unroll
  for (int i = 0; i < 2; ++i) {
#pragma unroll
    for (int j = 0; j < 4; ++j) {
      const int col = bn * 128 + wn * 64 + j * 16 + l16;
      const float bv = bias[col];
#pragma unroll
      for (int r = 0; r < 8; ++r) {
        const int row = bm * 128 + wm * 32 + i * 16 + half * 8 + r;
        const float v = acc[i][j][r] + bv;
        if (OUT_MODE == 0) {
          const int b  = row >> 11;
          const int s  = row & (Ss - 1);
          const int hh = col >> 6;
          const int d  = col & 63;
          ((__bf16*)Out)[(((size_t)b * Hh + hh) * Ss + s) * DH + d] = f2bf(v);
        } else {
          ((float*)Out)[(size_t)row * N + col] = v;
        }
      }
    }
  }
}

// ---------------------------------------------------------------------------
// Flash attention (causal): WG per (128-row q-tile, head, batch). 8 waves;
// wave w owns q-rows [qtile*128 + w*16, +16). K tile via ASYNC global->LDS,
// V^T tile staged with packed b32 stores.
// ---------------------------------------------------------------------------
__global__ __launch_bounds__(256) void flash_attn(
    const __bf16* __restrict__ Qg, const __bf16* __restrict__ Kg,
    const __bf16* __restrict__ Vg, __bf16* __restrict__ Og) {
  __shared__ alignas(16) __bf16 lK[128 * 64];      // [kpos][d]   16 KB
  __shared__ alignas(16) __bf16 lVt[64 * 128];     // [d][kpos]   16 KB
  __shared__ alignas(16) __bf16 lP[8 * 16 * 128];  // per-wave P  32 KB

  const int qtile = blockIdx.x, h = blockIdx.y, b = blockIdx.z;
  const int tid  = threadIdx.x;
  const int wave = tid >> 5, lane = tid & 31;
  const int half = lane >> 4, l16 = lane & 15;

  const size_t headoff = ((size_t)b * Hh + h) * Ss * DH;

  // Q fragments (A-layout, 2 chunks of k=32)
  const __bf16* qg = Qg + headoff + (size_t)(qtile * 128 + wave * 16) * DH;
  v16bf qfr[2];
#pragma unroll
  for (int c = 0; c < 2; ++c) {
    const __bf16* p = qg + l16 * DH + c * 32 + half * 8;
    ((v8bf*)&qfr[c])[0] = *(const v8bf*)p;
    ((v8bf*)&qfr[c])[1] = *(const v8bf*)(p + 16);
  }

  v8f accO[4] = {};
  float mrow[8], lrow[8];
#pragma unroll
  for (int r = 0; r < 8; ++r) { mrow[r] = -1e30f; lrow[r] = 0.f; }

  __bf16* pw = &lP[wave * 16 * 128];
  const int crow = tid >> 1;        // K staging row 0..127
  const int cd   = (tid & 1) * 32;  // K staging d offset 0/32
  const int vkp  = (tid >> 2) * 2;  // V staging kpos pair 0..126
  const int vd   = (tid & 3) * 16;  // V staging d base 0..48

  for (int kb = 0; kb <= qtile; ++kb) {
    // ---- stage K tile [kpos][d]: async global->LDS when available ----
    {
      const __bf16* gk = Kg + headoff + (size_t)(kb * 128 + crow) * DH + cd;
#if HAS_ASYNC_LDS
#pragma unroll
      for (int j = 0; j < 4; ++j)
        async_copy16(gk + j * 8, &lK[crow * 64 + cd + j * 8]);
#else
#pragma unroll
      for (int j = 0; j < 4; ++j)
        *(v8bf*)&lK[crow * 64 + cd + j * 8] = *(const v8bf*)(gk + j * 8);
#endif
    }
    // ---- stage V tile transposed [d][kpos], packed b32 stores ----
    {
      const __bf16* gv = Vg + headoff + (size_t)(kb * 128 + vkp) * DH + vd;
      v8bf r0a = *(const v8bf*)gv,        r0b = *(const v8bf*)(gv + 8);
      v8bf r1a = *(const v8bf*)(gv + DH), r1b = *(const v8bf*)(gv + DH + 8);
#pragma unroll
      for (int j = 0; j < 8; ++j) {
        v2bf p;
        p[0] = r0a[j];
        p[1] = r1a[j];
        *(v2bf*)&lVt[(vd + j) * 128 + vkp] = p;
      }
#pragma unroll
      for (int j = 0; j < 8; ++j) {
        v2bf p;
        p[0] = r0b[j];
        p[1] = r1b[j];
        *(v2bf*)&lVt[(vd + 8 + j) * 128 + vkp] = p;
      }
    }
#if HAS_ASYNC_LDS
    async_wait0();
#endif
    __syncthreads();

    // ---- S = Q K^T : 16 WMMA ----
    v8f sc[8] = {};
#pragma unroll
    for (int c = 0; c < 2; ++c) {
#pragma unroll
      for (int nf = 0; nf < 8; ++nf) {
        const __bf16* p = &lK[(nf * 16 + l16) * 64 + c * 32 + half * 8];
        v16bf bk;
        ((v8bf*)&bk)[0] = *(const v8bf*)p;
        ((v8bf*)&bk)[1] = *(const v8bf*)(p + 16);
        sc[nf] = wmma_bf16(qfr[c], bk, sc[nf]);
      }
    }

    // ---- causal mask + online softmax (rows r+8*half per VGPR) ----
    const int qrow0 = qtile * 128 + wave * 16 + half * 8;
    float nm[8];
#pragma unroll
    for (int r = 0; r < 8; ++r) nm[r] = mrow[r];
#pragma unroll
    for (int nf = 0; nf < 8; ++nf) {
      const int kpos = kb * 128 + nf * 16 + l16;
#pragma unroll
      for (int r = 0; r < 8; ++r) {
        float sv = sc[nf][r] * 0.125f;          // 1/sqrt(64)
        sv = (kpos <= qrow0 + r) ? sv : -1e30f;
        sc[nf][r] = sv;
        nm[r] = fmaxf(nm[r], sv);
      }
    }
#pragma unroll
    for (int dx = 1; dx < 16; dx <<= 1)
#pragma unroll
      for (int r = 0; r < 8; ++r)
        nm[r] = fmaxf(nm[r], __shfl_xor(nm[r], dx, 32));

    float corr[8], rs[8];
#pragma unroll
    for (int r = 0; r < 8; ++r) {
      corr[r] = __expf(mrow[r] - nm[r]);
      mrow[r] = nm[r];
      rs[r] = 0.f;
    }
#pragma unroll
    for (int j = 0; j < 4; ++j)
#pragma unroll
      for (int r = 0; r < 8; ++r) accO[j][r] *= corr[r];

#pragma unroll
    for (int nf = 0; nf < 8; ++nf)
#pragma unroll
      for (int r = 0; r < 8; ++r) {
        const float pv = __expf(sc[nf][r] - nm[r]);
        sc[nf][r] = pv;
        rs[r] += pv;
      }
#pragma unroll
    for (int dx = 1; dx < 16; dx <<= 1)
#pragma unroll
      for (int r = 0; r < 8; ++r) rs[r] += __shfl_xor(rs[r], dx, 32);
#pragma unroll
    for (int r = 0; r < 8; ++r) lrow[r] = lrow[r] * corr[r] + rs[r];

    // ---- P: C-layout -> per-wave LDS -> A-layout fragments ----
#pragma unroll
    for (int nf = 0; nf < 8; ++nf)
#pragma unroll
      for (int r = 0; r < 8; ++r)
        pw[(half * 8 + r) * 128 + nf * 16 + l16] = f2bf(sc[nf][r]);

    // ---- O += P V : 16 WMMA ----
#pragma unroll
    for (int c = 0; c < 4; ++c) {
      v16bf pf;
      const __bf16* pp = pw + l16 * 128 + c * 32 + half * 8;
      ((v8bf*)&pf)[0] = *(const v8bf*)pp;
      ((v8bf*)&pf)[1] = *(const v8bf*)(pp + 16);
#pragma unroll
      for (int j = 0; j < 4; ++j) {
        const __bf16* vp = &lVt[(j * 16 + l16) * 128 + c * 32 + half * 8];
        v16bf vf;
        ((v8bf*)&vf)[0] = *(const v8bf*)vp;
        ((v8bf*)&vf)[1] = *(const v8bf*)(vp + 16);
        accO[j] = wmma_bf16(pf, vf, accO[j]);
      }
    }
    __syncthreads();
  }

  // ---- normalize and write O[B*S, H*64] bf16 ----
#pragma unroll
  for (int r = 0; r < 8; ++r) {
    const float inv = 1.0f / lrow[r];
    const int srow = qtile * 128 + wave * 16 + half * 8 + r;
#pragma unroll
    for (int j = 0; j < 4; ++j) {
      const int col = h * DH + j * 16 + l16;
      Og[((size_t)b * Ss + srow) * (Hh * DH) + col] = f2bf(accO[j][r] * inv);
    }
  }
}

// ---------------------------------------------------------------------------
extern "C" void kernel_launch(void* const* d_in, const int* in_sizes, int n_in,
                              void* d_out, int out_size, void* d_ws, size_t ws_size,
                              hipStream_t stream) {
  const float* x   = (const float*)d_in[0];
  const float* w_q = (const float*)d_in[1];
  const float* b_q = (const float*)d_in[2];
  const float* w_k = (const float*)d_in[3];
  const float* b_k = (const float*)d_in[4];
  const float* w_v = (const float*)d_in[5];
  const float* b_v = (const float*)d_in[6];
  const float* w_o = (const float*)d_in[7];
  const float* b_o = (const float*)d_in[8];

  const size_t per = (size_t)Bb * Hh * Ss * DH;
  __bf16* Q = (__bf16*)d_ws;
  __bf16* K = Q + per;
  __bf16* V = K + per;
  __bf16* O = V + per;

  dim3 blk(256);
  dim3 gproj(Dd / 128, Mm / 128);
  dim3 gatt(Ss / 128, Hh, Bb);

  hipLaunchKernelGGL((gemm128<0, 0>), gproj, blk, 0, stream, (const void*)x, w_q, b_q, (void*)Q, Mm, Dd, Dd);
  hipLaunchKernelGGL((gemm128<0, 0>), gproj, blk, 0, stream, (const void*)x, w_k, b_k, (void*)K, Mm, Dd, Dd);
  hipLaunchKernelGGL((gemm128<0, 0>), gproj, blk, 0, stream, (const void*)x, w_v, b_v, (void*)V, Mm, Dd, Dd);
  hipLaunchKernelGGL(flash_attn, gatt, blk, 0, stream, (const __bf16*)Q, (const __bf16*)K, (const __bf16*)V, O);
  hipLaunchKernelGGL((gemm128<1, 1>), gproj, blk, 0, stream, (const void*)O, w_o, b_o, d_out, Mm, Dd, Dd);
}
